// DeepBeamformer_20229295964473
// MI455X (gfx1250) — matrix operand order repaired and checked by reading, not compiled, not run on hardware
//
#include <hip/hip_runtime.h>
#include <math.h>

// ---------------------------------------------------------------------------
// DeepBeamformer forward for MI455X (gfx1250, wave32, WMMA).
// Pipeline: conv1..3 -> primary caps + squash -> 3x dynamic routing
// (softmax / WMMA GEMM s=c*u_hat folded / squash / logit update) ->
// length + peak top-4 -> masked MLP (WMMA GEMM chain, fused GELU) -> out.
// ---------------------------------------------------------------------------

typedef __attribute__((ext_vector_type(16))) _Float16 v16h;
typedef __attribute__((ext_vector_type(8)))  float    v8f;

union H8 { uint4 q; _Float16 h[8]; };

__device__ __forceinline__ float gelu_exact(float x) {
    return 0.5f * x * (1.0f + erff(x * 0.70710678118654752f));
}

// ---------------- conv stack (VALU, L2-resident, ~3 GMAC total) ------------

__global__ void k_conv1(const float* __restrict__ in, const float* __restrict__ w,
                        const float* __restrict__ bias, float* __restrict__ out) {
    int b = blockIdx.x, tid = threadIdx.x;
    __shared__ float sm[3 * 64];
    for (int t = tid; t < 192; t += blockDim.x) sm[t] = in[(size_t)b * 192 + t];
    __syncthreads();
    for (int t = tid; t < 64 * 49; t += blockDim.x) {
        int oc = t / 49, p = t % 49, y = p / 7, x = p % 7;
        float acc = bias[oc];
        for (int ic = 0; ic < 3; ++ic) {
            const float* ip = sm + ic * 64 + y * 8 + x;
            const float* wp = w + ((size_t)oc * 3 + ic) * 4;
            acc += ip[0] * wp[0] + ip[1] * wp[1] + ip[8] * wp[2] + ip[9] * wp[3];
        }
        out[(size_t)b * 3136 + t] = gelu_exact(acc);
    }
}

__global__ void k_conv2(const float* __restrict__ in, const float* __restrict__ w,
                        const float* __restrict__ bias, float* __restrict__ out) {
    int b = blockIdx.x, tid = threadIdx.x;
    __shared__ float sm[64 * 49];
    for (int t = tid; t < 3136; t += blockDim.x) sm[t] = in[(size_t)b * 3136 + t];
    __syncthreads();
    for (int t = tid; t < 128 * 36; t += blockDim.x) {
        int oc = t / 36, p = t % 36, y = p / 6, x = p % 6;
        float acc = bias[oc];
        const float* wb = w + (size_t)oc * 64 * 4;
        for (int ic = 0; ic < 64; ++ic) {
            const float* ip = sm + ic * 49 + y * 7 + x;
            const float* wp = wb + ic * 4;
            acc += ip[0] * wp[0] + ip[1] * wp[1] + ip[7] * wp[2] + ip[8] * wp[3];
        }
        out[(size_t)b * 4608 + t] = gelu_exact(acc);
    }
}

__global__ void k_conv3(const float* __restrict__ in, const float* __restrict__ w,
                        const float* __restrict__ bias, float* __restrict__ out) {
    int b = blockIdx.x, tid = threadIdx.x;
    __shared__ float sm[128 * 36];
    for (int t = tid; t < 4608; t += blockDim.x) sm[t] = in[(size_t)b * 4608 + t];
    __syncthreads();
    for (int t = tid; t < 256 * 25; t += blockDim.x) {
        int oc = t / 25, p = t % 25, y = p / 5, x = p % 5;
        float acc = bias[oc];
        const float* wb = w + (size_t)oc * 128 * 4;
        for (int ic = 0; ic < 128; ++ic) {
            const float* ip = sm + ic * 36 + y * 6 + x;
            const float* wp = wb + ic * 4;
            acc += ip[0] * wp[0] + ip[1] * wp[1] + ip[6] * wp[2] + ip[7] * wp[3];
        }
        out[(size_t)b * 6400 + t] = gelu_exact(acc);
    }
}

// primary caps conv (256->96, k=2, 5x5->4x4) + squash; emits u in f32 and f16
__global__ void k_pc(const float* __restrict__ in, const float* __restrict__ w,
                     const float* __restrict__ bias,
                     float* __restrict__ u32, _Float16* __restrict__ u16) {
    int b = blockIdx.x, tid = threadIdx.x;
    __shared__ float sm[256 * 25];
    __shared__ float pcout[96 * 16];
    for (int t = tid; t < 6400; t += blockDim.x) sm[t] = in[(size_t)b * 6400 + t];
    __syncthreads();
    for (int t = tid; t < 96 * 16; t += blockDim.x) {
        int oc = t / 16, p = t % 16, y = p / 4, x = p % 4;
        float acc = bias[oc];
        const float* wb = w + (size_t)oc * 256 * 4;
        for (int ic = 0; ic < 256; ++ic) {
            const float* ip = sm + ic * 25 + y * 5 + x;
            const float* wp = wb + ic * 4;
            acc += ip[0] * wp[0] + ip[1] * wp[1] + ip[5] * wp[2] + ip[6] * wp[3];
        }
        pcout[t] = acc;
    }
    __syncthreads();
    for (int i = tid; i < 96; i += blockDim.x) {
        float ss = 0.f;
        for (int d = 0; d < 16; ++d) { float v = pcout[i * 16 + d]; ss += v * v; }
        float n = sqrtf(ss);
        float sc = (1.0f - expf(-n)) / (n + 1e-8f);
        for (int d = 0; d < 16; ++d) {
            float v = pcout[i * 16 + d] * sc;
            u32[((size_t)b * 96 + i) * 16 + d] = v;
            u16[((size_t)b * 96 + i) * 16 + d] = (_Float16)v;
        }
    }
}

// ---------------- weight format conversion ---------------------------------

__global__ void k_f2h(const float* __restrict__ src, _Float16* __restrict__ dst, int n) {
    int i = blockIdx.x * blockDim.x + threadIdx.x;
    if (i < n) dst[i] = (_Float16)src[i];
}

// caps_W [o,i,d,k] f32 -> W16 [o, (i*16+k), d] f16 (B-operand layout: row=K dim)
__global__ void k_capsW16(const float* __restrict__ capsW, _Float16* __restrict__ W16) {
    int i = blockIdx.x * blockDim.x + threadIdx.x;
    const int total = 181 * 96 * 256;
    if (i >= total) return;
    int k = i & 15, d = (i >> 4) & 15, oi = i >> 8;
    W16[(size_t)oi * 256 + k * 16 + d] = (_Float16)capsW[i];
}

__global__ void k_init_blog(const float* __restrict__ caps_b, float* __restrict__ blog) {
    size_t i = (size_t)blockIdx.x * blockDim.x + threadIdx.x;
    const size_t total = (size_t)512 * 181 * 96;
    if (i >= total) return;
    blog[i] = caps_b[i % (181 * 96)];
}

// ---------------- routing --------------------------------------------------

// softmax over out-caps dim o (torch dim=1); layouts [b][o][i]
__global__ void k_softmax(const float* __restrict__ blog, float* __restrict__ cbuf) {
    int b = blockIdx.x, i = threadIdx.x;            // blockDim = 96
    const float* p = blog + (size_t)b * 181 * 96 + i;
    float mx = -1e30f;
    for (int o = 0; o < 181; ++o) mx = fmaxf(mx, p[o * 96]);
    float s = 0.f;
    for (int o = 0; o < 181; ++o) s += expf(p[o * 96] - mx);
    float inv = 1.0f / s;
    float* q = cbuf + (size_t)b * 181 * 96 + i;
    for (int o = 0; o < 181; ++o) q[o * 96] = expf(p[o * 96] - mx) * inv;
}

// s[b,o,d] = sum_{i,k} (c[b,o,i]*u[b,i,k]) * W[o,i,d,k]  via WMMA, fused squash.
// Grid: (32 batch tiles, 181 o). One wave per block.
__global__ __launch_bounds__(32)
void k_route_sv(const float* __restrict__ cbuf,     // [B,181,96]
                const _Float16* __restrict__ u16,   // [B,1536]
                const _Float16* __restrict__ W16,   // [181,1536,16]
                float* __restrict__ vbuf) {         // [B,181,16]
    const int o = blockIdx.y;
    const int btile = blockIdx.x;
    const int lane = threadIdx.x;
    const int mrow = lane & 15;
    const int bglob = btile * 16 + mrow;
    const int base = (lane >> 4) * 8;               // A layout: K half-select
    const _Float16* urow = u16 + (size_t)bglob * 1536;
    const float*    crow = cbuf + ((size_t)bglob * 181 + o) * 96;
    const _Float16* wmat = W16 + (size_t)o * 1536 * 16;
    v8f acc = {};
    for (int kk = 0; kk < 48; ++kk) {
        int k0 = kk * 32;
        v16h a, bm;
        #pragma unroll
        for (int g = 0; g < 2; ++g) {               // A: lane M=lane%16, K runs of 8
            int col = k0 + g * 16 + base;
            H8 hu; hu.q = *(const uint4*)(urow + col);
            _Float16 ch = (_Float16)crow[col >> 4]; // same i for whole 8-run
            #pragma unroll
            for (int e = 0; e < 8; ++e) a[g * 8 + e] = hu.h[e] * ch;
        }
        const _Float16* wr = wmat + (lane & 15);  // B: lane%16 = N col, K runs of 8
        #pragma unroll
        for (int e = 0; e < 8; ++e) { bm[e] = wr[(size_t)(k0 + base + e) * 16]; bm[8 + e] = wr[(size_t)(k0 + 16 + base + e) * 16]; }
        acc = __builtin_amdgcn_wmma_f32_16x16x32_f16(false, a, false, bm,
                                                     (short)0, acc, false, false);
    }
    // fused squash: D layout lane%16 = d, row = j + 8*(lane>=16)
    int halfbase = (lane >= 16) ? 8 : 0;
    int d = lane & 15;
    #pragma unroll
    for (int j = 0; j < 8; ++j) {
        float val = acc[j];
        float ss = val * val;
        ss += __shfl_xor(ss, 1, 32);
        ss += __shfl_xor(ss, 2, 32);
        ss += __shfl_xor(ss, 4, 32);
        ss += __shfl_xor(ss, 8, 32);
        float n = sqrtf(ss);
        float sc = (1.0f - expf(-n)) / (n + 1e-8f);
        int brow = btile * 16 + j + halfbase;
        vbuf[((size_t)brow * 181 + o) * 16 + d] = val * sc;
    }
}

// b[b,o,i] += sum_d v[b,o,d] * (sum_k W[o,i,d,k] * u[b,i,k])
__global__ void k_bupdate(const float* __restrict__ vbuf, const float* __restrict__ capsW,
                          const float* __restrict__ u32, float* __restrict__ blog) {
    int o = blockIdx.x, b = blockIdx.y, i = threadIdx.x;   // blockDim = 96
    __shared__ float vsh[16];
    if (i < 16) vsh[i] = vbuf[((size_t)b * 181 + o) * 16 + i];
    __syncthreads();
    float ur[16];
    const float* up = u32 + ((size_t)b * 96 + i) * 16;
    #pragma unroll
    for (int k = 0; k < 16; ++k) ur[k] = up[k];
    const float* Wp = capsW + ((size_t)(o * 96 + i)) * 256;
    float accum = 0.f;
    for (int d = 0; d < 16; ++d) {
        const float* wd = Wp + d * 16;
        float s = 0.f;
        #pragma unroll
        for (int k = 0; k < 16; ++k) s += wd[k] * ur[k];
        accum += vsh[d] * s;
    }
    blog[((size_t)b * 181 + o) * 96 + i] += accum;
}

// ---------------- length / peaks -------------------------------------------

__global__ void k_length(const float* __restrict__ vbuf, float* __restrict__ len_out,
                         float* __restrict__ lenws) {
    int b = blockIdx.x, tid = threadIdx.x;                 // blockDim = 256
    __shared__ float nrm[181];
    __shared__ float red[256];
    float myn = 0.f;
    if (tid < 181) {
        const float* p = vbuf + ((size_t)b * 181 + tid) * 16;
        float ss = 0.f;
        #pragma unroll
        for (int d = 0; d < 16; ++d) ss += p[d] * p[d];
        myn = sqrtf(ss);
        nrm[tid] = myn;
    }
    red[tid] = (tid < 181) ? myn : 0.0f;
    __syncthreads();
    for (int s = 128; s > 0; s >>= 1) {
        if (tid < s) red[tid] += red[tid + s];
        __syncthreads();
    }
    float inv = 1.0f / (red[0] + 1e-8f);
    if (tid < 181) {
        float l = nrm[tid] * inv;
        len_out[(size_t)b * 181 + tid] = l;
        lenws[(size_t)b * 181 + tid] = l;
    }
}

// max-pool window 11, keep local maxima, top-4 (ties: lowest index), sort idx
__global__ void k_peaks(const float* __restrict__ lenws, int* __restrict__ idxbuf) {
    int b = blockIdx.x * blockDim.x + threadIdx.x;
    if (b >= 512) return;
    const float* L = lenws + (size_t)b * 181;
    float vals[4] = {-1e30f, -1e30f, -1e30f, -1e30f};
    int ids[4] = {0, 0, 0, 0};
    for (int o = 0; o < 181; ++o) {
        int lo = o - 5 < 0 ? 0 : o - 5, hi = o + 5 > 180 ? 180 : o + 5;
        float m = -1e30f;
        for (int j = lo; j <= hi; ++j) m = fmaxf(m, L[j]);
        float pv = (L[o] == m) ? L[o] : 0.0f;
        if (pv > vals[3]) {
            int pos = 3;
            while (pos > 0 && pv > vals[pos - 1]) pos--;
            for (int j = 3; j > pos; --j) { vals[j] = vals[j - 1]; ids[j] = ids[j - 1]; }
            vals[pos] = pv; ids[pos] = o;
        }
    }
    for (int a = 0; a < 3; ++a)
        for (int c = 0; c < 3 - a; ++c)
            if (ids[c] > ids[c + 1]) { int t = ids[c]; ids[c] = ids[c + 1]; ids[c + 1] = t; }
    for (int j = 0; j < 4; ++j) idxbuf[b * 4 + j] = ids[j];
}

// ---------------- masked MLP -----------------------------------------------

// Layer 1: one-hot mask => only the 16-row slice of fc1_w contributes.
__global__ void k_mlp1(const float* __restrict__ vbuf, const int* __restrict__ idxbuf,
                       const float* __restrict__ w1, const float* __restrict__ b1,
                       _Float16* __restrict__ act) {
    int bk = blockIdx.x;                                   // 0..2047 (b*4+kslot)
    int b = bk >> 2, ks = bk & 3;
    int id = idxbuf[b * 4 + ks];
    __shared__ float xr[16];
    if (threadIdx.x < 16) xr[threadIdx.x] = vbuf[((size_t)b * 181 + id) * 16 + threadIdx.x];
    __syncthreads();
    for (int n = threadIdx.x; n < 1024; n += blockDim.x) {
        float acc = b1[n];
        const float* wc = w1 + (size_t)id * 16 * 1024 + n;
        #pragma unroll
        for (int d = 0; d < 16; ++d) acc += xr[d] * wc[(size_t)d * 1024];
        act[(size_t)bk * 1024 + n] = (_Float16)gelu_exact(acc);
    }
}

// Generic WMMA GEMM: C[M,N] = A[M,K] * B[K,N] + bias, optional GELU.
// mode 0: f16 out [M,N]; mode 1: head -> d_out as [B, 2A=128, K=4] transpose.
__global__ __launch_bounds__(32)
void k_gemm_wmma(const _Float16* __restrict__ A, const _Float16* __restrict__ Bw,
                 const float* __restrict__ bias, _Float16* __restrict__ out16,
                 float* __restrict__ outW, int K, int N, int gelu, int mode) {
    int mtile = blockIdx.x, ntile = blockIdx.y;
    int lane = threadIdx.x;
    int mrow = mtile * 16 + (lane & 15);
    int base = (lane >> 4) * 8;
    const _Float16* arow = A + (size_t)mrow * K;
    v8f acc = {};
    for (int k0 = 0; k0 < K; k0 += 32) {
        v16h a, bm;
        #pragma unroll
        for (int g = 0; g < 2; ++g) {
            H8 ha; ha.q = *(const uint4*)(arow + k0 + g * 16 + base);
            #pragma unroll
            for (int e = 0; e < 8; ++e) a[g * 8 + e] = ha.h[e];
        }
        const _Float16* br = Bw + ntile * 16 + (lane & 15);
        #pragma unroll
        for (int e = 0; e < 8; ++e) { bm[e] = br[(size_t)(k0 + base + e) * N]; bm[8 + e] = br[(size_t)(k0 + 16 + base + e) * N]; }
        acc = __builtin_amdgcn_wmma_f32_16x16x32_f16(false, a, false, bm,
                                                     (short)0, acc, false, false);
    }
    int n = ntile * 16 + (lane & 15);
    int mb = (lane >= 16) ? 8 : 0;
    float bv = bias[n];
    #pragma unroll
    for (int j = 0; j < 8; ++j) {
        int m = mtile * 16 + j + mb;
        float v = acc[j] + bv;
        if (gelu) v = gelu_exact(v);
        if (mode == 0) {
            out16[(size_t)m * N + n] = (_Float16)v;
        } else {
            int bidx = m >> 2, kslot = m & 3;          // W_out[b, n, kslot]
            outW[(size_t)bidx * (128 * 4) + n * 4 + kslot] = v;
        }
    }
}

// ---------------------------------------------------------------------------

extern "C" void kernel_launch(void* const* d_in, const int* in_sizes, int n_in,
                              void* d_out, int out_size, void* d_ws, size_t ws_size,
                              hipStream_t stream) {
    (void)in_sizes; (void)n_in; (void)out_size; (void)ws_size;
    const float* scm   = (const float*)d_in[0];
    // d_in[1] = K (always 4 in reference setup)
    const float* c1w = (const float*)d_in[2];  const float* c1b = (const float*)d_in[3];
    const float* c2w = (const float*)d_in[4];  const float* c2b = (const float*)d_in[5];
    const float* c3w = (const float*)d_in[6];  const float* c3b = (const float*)d_in[7];
    const float* pcw = (const float*)d_in[8];  const float* pcb = (const float*)d_in[9];
    const float* capsW = (const float*)d_in[10]; const float* capsb = (const float*)d_in[11];
    const float* w1 = (const float*)d_in[12];  const float* b1 = (const float*)d_in[13];
    const float* w2 = (const float*)d_in[14];  const float* b2 = (const float*)d_in[15];
    const float* w3 = (const float*)d_in[16];  const float* b3 = (const float*)d_in[17];
    const float* w4 = (const float*)d_in[18];  const float* b4 = (const float*)d_in[19];
    const float* w5 = (const float*)d_in[20];  const float* b5 = (const float*)d_in[21];
    const float* wh = (const float*)d_in[22];  const float* bh = (const float*)d_in[23];
    float* out = (float*)d_out;
    float* out_len = out + (size_t)512 * 128 * 4;   // length follows W_out

    char* p = (char*)d_ws;
    auto alloc = [&](size_t bytes) -> char* {
        char* r = p; p += (bytes + 255) & ~(size_t)255; return r;
    };
    float*    h1    = (float*)alloc((size_t)512 * 3136 * 4);
    float*    h2    = (float*)alloc((size_t)512 * 4608 * 4);
    float*    h3    = (float*)alloc((size_t)512 * 6400 * 4);
    float*    u32   = (float*)alloc((size_t)512 * 1536 * 4);
    _Float16* u16   = (_Float16*)alloc((size_t)512 * 1536 * 2);
    _Float16* W16   = (_Float16*)alloc((size_t)181 * 1536 * 16 * 2);
    float*    blog  = (float*)alloc((size_t)512 * 181 * 96 * 4);
    float*    cbuf  = (float*)alloc((size_t)512 * 181 * 96 * 4);
    float*    vbuf  = (float*)alloc((size_t)512 * 181 * 16 * 4);
    float*    lenws = (float*)alloc((size_t)512 * 181 * 4);
    int*      idxb  = (int*)alloc((size_t)512 * 4 * 4);
    _Float16* actA  = (_Float16*)alloc((size_t)2048 * 1024 * 2);
    _Float16* actB  = (_Float16*)alloc((size_t)2048 * 1024 * 2);
    _Float16* w2h   = (_Float16*)alloc((size_t)1024 * 768 * 2);
    _Float16* w3h   = (_Float16*)alloc((size_t)768 * 512 * 2);
    _Float16* w4h   = (_Float16*)alloc((size_t)512 * 512 * 2);
    _Float16* w5h   = (_Float16*)alloc((size_t)512 * 256 * 2);
    _Float16* whh   = (_Float16*)alloc((size_t)256 * 128 * 2);

    // weight conversions
    {
        int n = 181 * 96 * 256;
        k_capsW16<<<(n + 255) / 256, 256, 0, stream>>>(capsW, W16);
        n = 1024 * 768; k_f2h<<<(n + 255) / 256, 256, 0, stream>>>(w2, w2h, n);
        n = 768 * 512;  k_f2h<<<(n + 255) / 256, 256, 0, stream>>>(w3, w3h, n);
        n = 512 * 512;  k_f2h<<<(n + 255) / 256, 256, 0, stream>>>(w4, w4h, n);
        n = 512 * 256;  k_f2h<<<(n + 255) / 256, 256, 0, stream>>>(w5, w5h, n);
        n = 256 * 128;  k_f2h<<<(n + 255) / 256, 256, 0, stream>>>(wh, whh, n);
    }

    // conv stack + primary capsules
    k_conv1<<<512, 256, 0, stream>>>(scm, c1w, c1b, h1);
    k_conv2<<<512, 256, 0, stream>>>(h1, c2w, c2b, h2);
    k_conv3<<<512, 256, 0, stream>>>(h2, c3w, c3b, h3);
    k_pc<<<512, 256, 0, stream>>>(h3, pcw, pcb, u32, u16);

    // routing
    {
        size_t tot = (size_t)512 * 181 * 96;
        k_init_blog<<<(int)((tot + 255) / 256), 256, 0, stream>>>(capsb, blog);
    }
    for (int it = 0; it < 3; ++it) {
        k_softmax<<<512, 96, 0, stream>>>(blog, cbuf);
        k_route_sv<<<dim3(32, 181), 32, 0, stream>>>(cbuf, u16, W16, vbuf);
        if (it < 2)
            k_bupdate<<<dim3(181, 512), 96, 0, stream>>>(vbuf, capsW, u32, blog);
    }

    // length + peaks
    k_length<<<512, 256, 0, stream>>>(vbuf, out_len, lenws);
    k_peaks<<<(512 + 63) / 64, 64, 0, stream>>>(lenws, idxb);

    // masked MLP
    k_mlp1<<<2048, 256, 0, stream>>>(vbuf, idxb, w1, b1, actA);
    k_gemm_wmma<<<dim3(128, 48), 32, 0, stream>>>(actA, w2h, b2, actB, nullptr, 1024, 768, 1, 0);
    k_gemm_wmma<<<dim3(128, 32), 32, 0, stream>>>(actB, w3h, b3, actA, nullptr, 768, 512, 1, 0);
    k_gemm_wmma<<<dim3(128, 32), 32, 0, stream>>>(actA, w4h, b4, actB, nullptr, 512, 512, 1, 0);
    k_gemm_wmma<<<dim3(128, 16), 32, 0, stream>>>(actB, w5h, b5, actA, nullptr, 512, 256, 1, 0);
    k_gemm_wmma<<<dim3(128, 8),  32, 0, stream>>>(actA, whh, bh, nullptr, out, 256, 128, 0, 1);
}
